// Model_9242769621764
// MI455X (gfx1250) — matrix operand order repaired and checked by reading, not compiled, not run on hardware
//
#include <hip/hip_runtime.h>
#include <hip/hip_bf16.h>
#include <cstdint>

typedef __attribute__((ext_vector_type(16))) _Float16 v16h;
typedef __attribute__((ext_vector_type(8)))  _Float16 v8h;
typedef __attribute__((ext_vector_type(8)))  float    v8f;
typedef __attribute__((ext_vector_type(4)))  unsigned int u32x4;
typedef __attribute__((ext_vector_type(8)))  int          i32x8;
typedef __attribute__((ext_vector_type(4)))  int          i32x4;

#define K_NEI 32

#if defined(__has_builtin)
#if __has_builtin(__builtin_amdgcn_tensor_load_to_lds) && \
    __has_builtin(__builtin_amdgcn_s_wait_tensorcnt)
#define HAVE_TDM 1
#endif
#endif

// ---------------------------------------------------------------------------
// Weight conversion: f32 [Cout][Cin] -> f16 [Mpad][Kpad], zero padded.
// ---------------------------------------------------------------------------
__global__ void cvt_weight(const float* __restrict__ src, _Float16* __restrict__ dst,
                           int Cout, int Cin, int Mpad, int Kpad) {
  long i = (long)blockIdx.x * blockDim.x + threadIdx.x;
  long tot = (long)Mpad * Kpad;
  if (i >= tot) return;
  int r = (int)(i / Kpad), c = (int)(i % Kpad);
  float v = (r < Cout && c < Cin) ? src[(long)r * Cin + c] : 0.f;
  dst[i] = (_Float16)v;
}

// ---------------------------------------------------------------------------
// Pack input x [B][3][N] -> xyz f32 [B*N][3] and embedding act f16 [B*N][32].
// ---------------------------------------------------------------------------
__global__ void pack_x(const float* __restrict__ x, float* __restrict__ xyz,
                       _Float16* __restrict__ act, int N) {
  long i = (long)blockIdx.x * blockDim.x + threadIdx.x;
  if (i >= (long)32 * N) return;
  long b = i / N, n = i % N;
  float vx = x[(b * 3 + 0) * N + n];
  float vy = x[(b * 3 + 1) * N + n];
  float vz = x[(b * 3 + 2) * N + n];
  xyz[i * 3 + 0] = vx; xyz[i * 3 + 1] = vy; xyz[i * 3 + 2] = vz;
  _Float16* row = act + i * 32;
  row[0] = (_Float16)vx; row[1] = (_Float16)vy; row[2] = (_Float16)vz;
  for (int c = 3; c < 32; c++) row[c] = (_Float16)0.f;
}

// ---------------------------------------------------------------------------
// Direct-global WMMA GEMM (used for tiny P, e.g. classifier head):
// block M=64 x N=256, wave tile 64x32.
// ---------------------------------------------------------------------------
__global__ __launch_bounds__(256)
void wmma_gemm(const _Float16* __restrict__ W, const _Float16* __restrict__ X,
               float* __restrict__ Y, long P, int Kpad, int Mpad) {
  const int wave = threadIdx.x >> 5;
  const int lane = threadIdx.x & 31;
  const int lh   = lane & 15;
  const bool hi  = lane >= 16;
  const long n0  = (long)blockIdx.x * 256 + wave * 32;
  const int  m0  = blockIdx.y * 64;

  v8f acc[4][2];
#pragma unroll
  for (int mf = 0; mf < 4; mf++)
#pragma unroll
    for (int nf = 0; nf < 2; nf++)
#pragma unroll
      for (int e = 0; e < 8; e++) acc[mf][nf][e] = 0.f;

  const _Float16* aBase = W + (long)(m0 + lh) * Kpad + (hi ? 16 : 0);
  long pt0 = n0 + lh;       if (pt0 >= P) pt0 = 0;
  long pt1 = n0 + 16 + lh;  if (pt1 >= P) pt1 = 0;
  const _Float16* b0 = X + pt0 * Kpad + (hi ? 16 : 0);
  const _Float16* b1 = X + pt1 * Kpad + (hi ? 16 : 0);

  for (int kk = 0; kk < Kpad; kk += 32) {
    v16h a[4], bf[2];
#pragma unroll
    for (int mf = 0; mf < 4; mf++) {
      const _Float16* ap = aBase + (long)mf * 16 * Kpad + kk;
      v8h lo = *(const v8h*)ap;
      v8h h8 = *(const v8h*)(ap + 8);
#pragma unroll
      for (int e = 0; e < 8; e++) { a[mf][e] = lo[e]; a[mf][8 + e] = h8[e]; }
    }
    bf[0] = *(const v16h*)(b0 + kk);
    bf[1] = *(const v16h*)(b1 + kk);
    __builtin_prefetch(b0 + kk + 64, 0, 3);   // near-scope streaming prefetch
#pragma unroll
    for (int mf = 0; mf < 4; mf++)
#pragma unroll
      for (int nf = 0; nf < 2; nf++)
        acc[mf][nf] = __builtin_amdgcn_wmma_f32_16x16x32_f16(
            false, a[mf], false, bf[nf], (short)0, acc[mf][nf], false, false);
  }

  long ptA = n0 + lh, ptB = n0 + 16 + lh;
  int co = m0 + (hi ? 8 : 0);
#pragma unroll
  for (int mf = 0; mf < 4; mf++) {
    if (ptA < P) *(v8f*)(Y + ptA * Mpad + co + mf * 16) = acc[mf][0];
    if (ptB < P) *(v8f*)(Y + ptB * Mpad + co + mf * 16) = acc[mf][1];
  }
}

// ---------------------------------------------------------------------------
// TDM descriptor build + issue: 2D tile (tile_dim0=32 elems of 2B, tile_dim1=512
// rows), row stride Kpad elems, LDS rows padded 16DW + 4DW -> 80B pitch.
// D# layout per CDNA5 ISA 8.3/8.4. OOB rows/cols are zero-filled by the TDM.
// This toolchain exposes the 6-arg builtin: (g0, g1, g2, g3, g_ext, cpol).
// ---------------------------------------------------------------------------
#if HAVE_TDM
__device__ __forceinline__ void tdm_load_tile(unsigned ldsOff, const _Float16* gaddr,
                                              long rowsRem, int colsRem, int kpad) {
  unsigned long long ga = (unsigned long long)(uintptr_t)gaddr;
  unsigned dim0 = (unsigned)colsRem;          // tensor_dim0 (remaining elems in row)
  unsigned dim1 = (unsigned)rowsRem;          // tensor_dim1 (remaining rows)
  u32x4 g0;
  g0[0] = 1u;                                              // count=1 user descriptor
  g0[1] = ldsOff;                                          // lds_addr (bytes)
  g0[2] = (unsigned)(ga & 0xffffffffu);                    // global_addr lo
  g0[3] = (unsigned)((ga >> 32) & 0x01ffffffu) | (2u << 30); // global_addr hi | type=2
  i32x8 g1;
  g1[0] = (int)((1u << 16)        // data_size = 2 bytes
              | (1u << 20)        // pad_enable
              | (3u << 22)        // pad_interval: 16 DWORDs
              | (3u << 25));      // pad_amount:   4 DWORDs -> 80B row pitch
  g1[1] = (int)((dim0 & 0xffffu) << 16);                         // tensor_dim0 lo16
  g1[2] = (int)(((dim0 >> 16) & 0xffffu) | ((dim1 & 0xffffu) << 16)); // dim0 hi|dim1 lo
  g1[3] = (int)(((dim1 >> 16) & 0xffffu) | (32u << 16));         // dim1 hi | tile_dim0=32
  g1[4] = (int)512;                                              // tile_dim1=512
  g1[5] = (int)(unsigned)kpad;                                   // tensor_dim0_stride lo32
  g1[6] = 0;                                                     // stride hi | dim1_stride lo
  g1[7] = 0;
  i32x4 z4; z4[0] = z4[1] = z4[2] = z4[3] = 0;                   // 2D: groups 2/3 unused
  i32x8 z8;
#pragma unroll
  for (int e = 0; e < 8; e++) z8[e] = 0;
  __builtin_amdgcn_tensor_load_to_lds(g0, g1, z4, z4, z8, 0);
}
#endif

// ---------------------------------------------------------------------------
// TDM-staged WMMA GEMM for large P: block M=64 x N=512, 8 waves of 64x64
// (16 WMMAs / K-step). B panel (512x32 f16) is DMA'd into LDS by the Tensor
// Data Mover, double-buffered on TENSORcnt + workgroup barriers. A fragments
// are direct-global (identical across waves -> WGP$ broadcast).
// ---------------------------------------------------------------------------
__global__ __launch_bounds__(256)
void wmma_gemm_tdm(const _Float16* __restrict__ W, const _Float16* __restrict__ X,
                   float* __restrict__ Y, long P, int Kpad, int Mpad) {
  __shared__ _Float16 sB[2][512 * 40];   // 80B row pitch (32 data + 8 pad halfs)
  const int wave = threadIdx.x >> 5;
  const int lane = threadIdx.x & 31;
  const int lh   = lane & 15;
  const bool hi  = lane >= 16;
  const long n0b = (long)blockIdx.x * 512;
  const int  m0  = blockIdx.y * 64;
  const int  nw  = wave * 64;

  v8f acc[4][4];
#pragma unroll
  for (int mf = 0; mf < 4; mf++)
#pragma unroll
    for (int nf = 0; nf < 4; nf++)
#pragma unroll
      for (int e = 0; e < 8; e++) acc[mf][nf][e] = 0.f;

  const _Float16* aBase = W + (long)(m0 + lh) * Kpad + (hi ? 16 : 0);
  const int nsteps = Kpad / 32;

#if HAVE_TDM
  if (wave == 0)
    tdm_load_tile((unsigned)(uintptr_t)&sB[0][0], X + n0b * Kpad,
                  P - n0b, Kpad, Kpad);
#endif

  for (int i = 0; i < nsteps; i++) {
    const int kk = i * 32;
    const int buf = i & 1;
#if HAVE_TDM
    if (wave == 0) __builtin_amdgcn_s_wait_tensorcnt((short)0);
    __syncthreads();                       // tile i visible to all waves
    if (wave == 0 && i + 1 < nsteps)
      tdm_load_tile((unsigned)(uintptr_t)&sB[(i + 1) & 1][0],
                    X + n0b * Kpad + (kk + 32), P - n0b, Kpad - (kk + 32), Kpad);
#else
    __syncthreads();
    for (int r = threadIdx.x; r < 512; r += 256) {      // cooperative fallback
      long pt = n0b + r; if (pt >= P) pt = n0b;          // dup row; stores guarded
      *(v16h*)&sB[buf][r * 40] = *(const v16h*)(X + pt * Kpad + kk);
    }
    __syncthreads();
#endif
    v16h a[4], bf[4];
#pragma unroll
    for (int mf = 0; mf < 4; mf++) {
      const _Float16* ap = aBase + (long)mf * 16 * Kpad + kk;
      v8h lo = *(const v8h*)ap;
      v8h h8 = *(const v8h*)(ap + 8);
#pragma unroll
      for (int e = 0; e < 8; e++) { a[mf][e] = lo[e]; a[mf][8 + e] = h8[e]; }
    }
#pragma unroll
    for (int nf = 0; nf < 4; nf++)
      bf[nf] = *(const v16h*)&sB[buf][(nw + nf * 16 + lh) * 40 + (hi ? 16 : 0)];
#pragma unroll
    for (int mf = 0; mf < 4; mf++)
#pragma unroll
      for (int nf = 0; nf < 4; nf++)
        acc[mf][nf] = __builtin_amdgcn_wmma_f32_16x16x32_f16(
            false, a[mf], false, bf[nf], (short)0, acc[mf][nf], false, false);
#if HAVE_TDM
    __syncthreads();                       // all reads of buf done before reuse
#endif
  }

  const int co = m0 + (hi ? 8 : 0);
#pragma unroll
  for (int nf = 0; nf < 4; nf++) {
    long pt = n0b + nw + nf * 16 + lh;
    if (pt < P) {
#pragma unroll
      for (int mf = 0; mf < 4; mf++)
        *(v8f*)(Y + pt * Mpad + co + mf * 16) = acc[mf][nf];
    }
  }
}

// ---------------------------------------------------------------------------
// BN batch stats: per-channel sum & sumsq over P points (atomic accumulate).
// ---------------------------------------------------------------------------
__global__ void bn_stats(const float* __restrict__ Y, float* __restrict__ stats,
                         long P, int Mpad) {
  __shared__ float ssm[256], sqm[256];
  int cl = threadIdx.x & 63, ps = threadIdx.x >> 6;
  int ch = blockIdx.y * 64 + cl;
  long p0 = (long)blockIdx.x * 1024;
  long pend = p0 + 1024; if (pend > P) pend = P;
  float s = 0.f, q = 0.f;
  for (long p = p0 + ps; p < pend; p += 4) {
    float v = Y[p * Mpad + ch];
    s += v; q += v * v;
  }
  ssm[threadIdx.x] = s; sqm[threadIdx.x] = q;
  __syncthreads();
  if (ps == 0) {
    for (int j = 1; j < 4; j++) { s += ssm[cl + 64 * j]; q += sqm[cl + 64 * j]; }
    atomicAdd(&stats[ch], s);
    atomicAdd(&stats[Mpad + ch], q);
  }
}

__global__ void bn_finalize(const float* __restrict__ stats,
                            const float* __restrict__ gamma,
                            const float* __restrict__ beta,
                            float* __restrict__ ssh, int C, int Mpad, float invN) {
  int ch = blockIdx.x * blockDim.x + threadIdx.x;
  if (ch >= Mpad) return;
  float sc = 0.f, sh = 0.f;
  if (ch < C) {
    float m = stats[ch] * invN;
    float v = stats[Mpad + ch] * invN - m * m;     // biased (training-mode) var
    sc = gamma[ch] * rsqrtf(v + 1e-5f);
    sh = beta[ch] - m * sc;                         // conv bias cancels in BN
  }
  ssh[ch] = sc; ssh[Mpad + ch] = sh;
}

__global__ void bn_apply(const float* __restrict__ Y, const float* __restrict__ ssh,
                         const _Float16* __restrict__ resid, _Float16* __restrict__ out,
                         long P, int Mpad, int relu) {
  long i = (long)blockIdx.x * blockDim.x + threadIdx.x;
  int nc = Mpad >> 3;
  long tot = P * nc;
  if (i >= tot) return;
  long pt = i / nc; int cg = (int)(i % nc) * 8;
  v8f y = *(const v8f*)(Y + pt * Mpad + cg);
  v8h o;
#pragma unroll
  for (int j = 0; j < 8; j++) {
    float f = y[j] * ssh[cg + j] + ssh[Mpad + cg + j];
    if (resid) f += (float)resid[pt * Mpad + cg + j];
    if (relu)  f = fmaxf(f, 0.f);
    o[j] = (_Float16)f;
  }
  *(v8h*)(out + pt * Mpad + cg) = o;
}

// ---------------------------------------------------------------------------
// Farthest point sampling: one workgroup per batch, xyz + running min-dist in
// LDS, argmax via packed (dist_bits<<32 | idx) LDS atomicMax (ds_max_u64).
// ---------------------------------------------------------------------------
__global__ __launch_bounds__(256)
void fps_kernel(const float* __restrict__ xyz, int* __restrict__ fidx, int N, int S) {
  __shared__ float sx[2048], sy[2048], sz[2048], sd[2048];
  __shared__ unsigned long long best;
  __shared__ int curFar;
  __shared__ float cx, cy, cz;
  int b = blockIdx.x, t = threadIdx.x;
  const float* base = xyz + (long)b * N * 3;
  for (int i = t; i < N; i += 256) {
    sx[i] = base[i * 3]; sy[i] = base[i * 3 + 1]; sz[i] = base[i * 3 + 2];
    sd[i] = 1e10f;
  }
  if (t == 0) curFar = 0;
  __syncthreads();
  for (int s = 0; s < S; s++) {
    if (t == 0) {
      fidx[(long)b * S + s] = curFar;
      cx = sx[curFar]; cy = sy[curFar]; cz = sz[curFar];
      best = 0ULL;
    }
    __syncthreads();
    unsigned long long loc = 0ULL;
    for (int i = t; i < N; i += 256) {
      float dx = sx[i] - cx, dy = sy[i] - cy, dz = sz[i] - cz;
      float d = dx * dx + dy * dy + dz * dz;
      float nd = fminf(sd[i], d); sd[i] = nd;
      unsigned long long pk =
          ((unsigned long long)__float_as_uint(nd) << 32) | (unsigned)i;
      if (pk > loc) loc = pk;
    }
    atomicMax(&best, loc);
    __syncthreads();
    if (t == 0) curFar = (int)(best & 0xffffffffULL);
    __syncthreads();
  }
}

__global__ void gather_centers(const float* __restrict__ xyz, const int* __restrict__ fidx,
                               const _Float16* __restrict__ feat, float* __restrict__ nxyz,
                               _Float16* __restrict__ cfeat, int N, int S, int Cpad) {
  int g = blockIdx.x; int b = g / S;
  long src = (long)b * N + fidx[g];
  if (threadIdx.x == 0) {
    nxyz[(long)g * 3 + 0] = xyz[src * 3 + 0];
    nxyz[(long)g * 3 + 1] = xyz[src * 3 + 1];
    nxyz[(long)g * 3 + 2] = xyz[src * 3 + 2];
  }
  for (int c = threadIdx.x * 8; c < Cpad; c += 64 * 8)
    *(v8h*)(cfeat + (long)g * Cpad + c) = *(const v8h*)(feat + src * Cpad + c);
}

// ---------------------------------------------------------------------------
// kNN (K=32): one thread per center, LDS-tiled candidates, worst-replacement.
// ---------------------------------------------------------------------------
__global__ __launch_bounds__(128)
void knn_kernel(const float* __restrict__ xyz, const float* __restrict__ nxyz,
                int* __restrict__ nidx, int N, int S) {
  __shared__ float tx[128], ty[128], tz[128];
  int b = blockIdx.y;
  int s = blockIdx.x * 128 + threadIdx.x;
  long g = (long)b * S + s;
  float cx = nxyz[g * 3], cy = nxyz[g * 3 + 1], cz = nxyz[g * 3 + 2];
  float bd[K_NEI]; int bi[K_NEI];
#pragma unroll
  for (int k = 0; k < K_NEI; k++) { bd[k] = 3e38f; bi[k] = 0; }
  float wv = 3e38f; int wp = 0;
  const float* base = xyz + (long)b * N * 3;
  for (int t0 = 0; t0 < N; t0 += 128) {
    __syncthreads();
    int i = t0 + threadIdx.x;
    if (i < N) {
      tx[threadIdx.x] = base[i * 3];
      ty[threadIdx.x] = base[i * 3 + 1];
      tz[threadIdx.x] = base[i * 3 + 2];
    }
    __syncthreads();
    int lim = (N - t0 < 128) ? (N - t0) : 128;
    for (int j = 0; j < lim; j++) {
      float dx = tx[j] - cx, dy = ty[j] - cy, dz = tz[j] - cz;
      float d = dx * dx + dy * dy + dz * dz;
      if (d < wv) {
        bd[wp] = d; bi[wp] = t0 + j;
        wv = bd[0]; wp = 0;
#pragma unroll
        for (int k = 1; k < K_NEI; k++)
          if (bd[k] > wv) { wv = bd[k]; wp = k; }
      }
    }
  }
  for (int k = 0; k < K_NEI; k++) nidx[g * K_NEI + k] = bi[k];
}

// ---------------------------------------------------------------------------
// Grouping pass 1: per-batch sum of squared group-centered values (for std).
// ---------------------------------------------------------------------------
__global__ __launch_bounds__(128)
void grp_sumsq(const _Float16* __restrict__ feat, const float* __restrict__ xyz,
               const int* __restrict__ nidx, float* __restrict__ bsum,
               int N, int S, int C, int Cpad, int D) {
  __shared__ int sid[K_NEI];
  __shared__ float red[128];
  long g = blockIdx.x; int b = (int)(g / S);
  if (threadIdx.x < K_NEI) sid[threadIdx.x] = nidx[g * K_NEI + threadIdx.x];
  __syncthreads();
  float q = 0.f;
  for (int d = threadIdx.x; d < D; d += 128) {
    float vv[K_NEI]; float s = 0.f;
#pragma unroll
    for (int k = 0; k < K_NEI; k++) {
      long src = (long)b * N + sid[k];
      float v = (d < C) ? (float)feat[src * Cpad + d] : xyz[src * 3 + (d - C)];
      vv[k] = v; s += v;
    }
    float m = s * (1.f / K_NEI);
#pragma unroll
    for (int k = 0; k < K_NEI; k++) { float c0 = vv[k] - m; q += c0 * c0; }
  }
  red[threadIdx.x] = q; __syncthreads();
  for (int o = 64; o > 0; o >>= 1) {
    if (threadIdx.x < o) red[threadIdx.x] += red[threadIdx.x + o];
    __syncthreads();
  }
  if (threadIdx.x == 0) atomicAdd(&bsum[b], red[0]);
}

__global__ void grp_std(const float* __restrict__ bsum, float* __restrict__ stdinv,
                        float cntm1) {
  int b = threadIdx.x;
  stdinv[b] = 1.f / (sqrtf(bsum[b] / cntm1) + 1e-5f);  // unbiased torch.std
}

// Grouping pass 2: emit f16 [B*S*K][Dpad] = [alpha*(c/std)+beta | center feat]
__global__ __launch_bounds__(128)
void grp_emit(const _Float16* __restrict__ feat, const float* __restrict__ xyz,
              const int* __restrict__ nidx, const _Float16* __restrict__ cfeat,
              const float* __restrict__ alpha, const float* __restrict__ beta,
              const float* __restrict__ stdinv, _Float16* __restrict__ out,
              int N, int S, int C, int Cpad, int D, int Dpad) {
  __shared__ int sid[K_NEI];
  long g = blockIdx.x; int b = (int)(g / S);
  if (threadIdx.x < K_NEI) sid[threadIdx.x] = nidx[g * K_NEI + threadIdx.x];
  __syncthreads();
  float inv = stdinv[b];
  _Float16* orow = out + g * (long)K_NEI * Dpad;
  for (int d = threadIdx.x; d < D; d += 128) {
    float vv[K_NEI]; float s = 0.f;
#pragma unroll
    for (int k = 0; k < K_NEI; k++) {
      long src = (long)b * N + sid[k];
      float v = (d < C) ? (float)feat[src * Cpad + d] : xyz[src * 3 + (d - C)];
      vv[k] = v; s += v;
    }
    float m = s * (1.f / K_NEI);
    float al = alpha[d], be = beta[d];
#pragma unroll
    for (int k = 0; k < K_NEI; k++)
      orow[(long)k * Dpad + d] = (_Float16)(al * ((vv[k] - m) * inv) + be);
  }
  for (int d = threadIdx.x; d < C; d += 128) {
    _Float16 v = cfeat[g * (long)Cpad + d];
    for (int k = 0; k < K_NEI; k++)
      orow[(long)k * Dpad + D + d] = v;
  }
}

// max-reduce Kred consecutive rows (group pool over K, global pool over S)
__global__ void pool_rows(const _Float16* __restrict__ in, _Float16* __restrict__ out,
                          long G, int Kred, int Cpad) {
  long i = (long)blockIdx.x * blockDim.x + threadIdx.x;
  int nc = Cpad >> 3;
  if (i >= G * nc) return;
  long g = i / nc; int cg = (int)(i % nc) * 8;
  const _Float16* base = in + g * (long)Kred * Cpad + cg;
  v8h m = *(const v8h*)base;
  for (int k = 1; k < Kred; k++) {
    v8h v = *(const v8h*)(base + (long)k * Cpad);
#pragma unroll
    for (int e = 0; e < 8; e++)
      if ((float)v[e] > (float)m[e]) m[e] = v[e];
  }
  *(v8h*)(out + g * Cpad + cg) = m;
}

__global__ void final_bias(const float* __restrict__ Y, const float* __restrict__ bias,
                           float* __restrict__ out, int Mpad) {
  int i = blockIdx.x * blockDim.x + threadIdx.x;
  if (i >= 32 * 40) return;
  int b = i / 40, o = i % 40;
  out[i] = Y[(long)b * Mpad + o] + bias[o];
}

// ===========================================================================
// Host orchestration
// ===========================================================================
extern "C" void kernel_launch(void* const* d_in, const int* in_sizes, int n_in,
                              void* d_out, int out_size, void* d_ws, size_t ws_size,
                              hipStream_t stream) {
  (void)in_sizes; (void)n_in; (void)out_size; (void)ws_size;
  const int B = 32;

  // Input index map (pytree alphabetical key order):
  //  0:x | cls.l1 W1 b2 beta3 gamma4 | cls.l2 W5 b6 beta7 gamma8 | cls.l3 W9 b10
  //  embedding W11 b12 beta13 gamma14
  //  stage s base=15+s*38: alpha+0 beta+1 | pos0.c1+2 pos0.c2+6 pos1.c1+10
  //  pos1.c2+14 | pre0.c1+18 pre0.c2+22 pre1.c1+26 pre1.c2+30 | transfer+34

  char* basep = (char*)d_ws;
  size_t off = 0;
  auto carve = [&](size_t bytes) -> char* {
    char* p = basep + off;
    off = (off + bytes + 255) & ~(size_t)255;
    return p;
  };

  _Float16* featBuf = (_Float16*)carve((size_t)B * 2048 * 64 * 2);
  _Float16* actG    = (_Float16*)carve((size_t)1048576 * 160 * 2);
  _Float16* actP    = (_Float16*)carve((size_t)1048576 * 128 * 2);
  _Float16* actT    = (_Float16*)carve((size_t)1048576 * 128 * 2);
  float*    Ybuf    = (float*)   carve((size_t)1048576 * 128 * 4);
  float*    xyzA    = (float*)   carve((size_t)B * 2048 * 3 * 4);
  float*    xyzB    = (float*)   carve((size_t)B * 1024 * 3 * 4);
  int*      fidx    = (int*)     carve((size_t)B * 1024 * 4);
  int*      nidx    = (int*)     carve((size_t)B * 1024 * K_NEI * 4);
  _Float16* cfeat   = (_Float16*)carve((size_t)B * 65536 * 2);
  float*    stats   = (float*)   carve(2 * 1024 * 4);
  float*    ssh     = (float*)   carve(2 * 1024 * 4);
  float*    bsum    = (float*)   carve(32 * 4);
  float*    stdinv  = (float*)   carve(32 * 4);
  _Float16* hbuf    = (_Float16*)carve((size_t)B * 1024 * 2);

  auto cvt = [&](int inIdx, int Cout, int Cin, int Mpad, int Kpad) -> _Float16* {
    _Float16* w = (_Float16*)carve((size_t)Mpad * Kpad * 2);
    long tot = (long)Mpad * Kpad;
    cvt_weight<<<(unsigned)((tot + 255) / 256), 256, 0, stream>>>(
        (const float*)d_in[inIdx], w, Cout, Cin, Mpad, Kpad);
    return w;
  };
  _Float16* Wemb = cvt(11, 64, 3, 64, 32);
  _Float16* Wt[4]; _Float16* Wpre[4][4]; _Float16* Wpos[4][4];
  for (int s = 0; s < 4; s++) {
    int last = 64 << s, outc = 128 << s;
    int Dtot = 2 * last + 3, Dpad = ((Dtot + 31) / 32) * 32;
    int sb = 15 + s * 38;
    Wt[s] = cvt(sb + 34, outc, Dtot, outc, Dpad);
    const int preo[4] = {18, 22, 26, 30}, poso[4] = {2, 6, 10, 14};
    for (int j = 0; j < 4; j++) {
      Wpre[s][j] = cvt(sb + preo[j], outc, outc, outc, outc);
      Wpos[s][j] = cvt(sb + poso[j], outc, outc, outc, outc);
    }
  }
  _Float16* Wl1 = cvt(1, 512, 1024, 512, 1024);
  _Float16* Wl2 = cvt(5, 256, 512, 256, 512);
  _Float16* Wl3 = cvt(9, 40, 256, 64, 256);

  // fused conv + training-mode BN (+resid,+relu); TDM-staged GEMM for large P
  auto convbn = [&](const _Float16* W16, const _Float16* Xin, _Float16* Aout,
                    const _Float16* resid, int relu, long P, int Kpad, int Mpad,
                    int C, const float* gamma, const float* beta) {
    if (P >= 512) {
      dim3 gg((unsigned)((P + 511) / 512), Mpad / 64);
      wmma_gemm_tdm<<<gg, 256, 0, stream>>>(W16, Xin, Ybuf, P, Kpad, Mpad);
    } else {
      dim3 gg((unsigned)((P + 255) / 256), Mpad / 64);
      wmma_gemm<<<gg, 256, 0, stream>>>(W16, Xin, Ybuf, P, Kpad, Mpad);
    }
    (void)hipMemsetAsync(stats, 0, sizeof(float) * 2 * Mpad, stream);
    dim3 gs((unsigned)((P + 1023) / 1024), Mpad / 64);
    bn_stats<<<gs, 256, 0, stream>>>(Ybuf, stats, P, Mpad);
    bn_finalize<<<(Mpad + 63) / 64, 64, 0, stream>>>(stats, gamma, beta, ssh, C,
                                                     Mpad, 1.f / (float)P);
    long tot = P * (Mpad >> 3);
    bn_apply<<<(unsigned)((tot + 255) / 256), 256, 0, stream>>>(
        Ybuf, ssh, resid, Aout, P, Mpad, relu);
  };

  // --- embedding ---
  {
    long tot = (long)B * 2048;
    pack_x<<<(unsigned)((tot + 255) / 256), 256, 0, stream>>>(
        (const float*)d_in[0], xyzA, actT, 2048);
  }
  convbn(Wemb, actT, featBuf, nullptr, 1, (long)B * 2048, 32, 64, 64,
         (const float*)d_in[14], (const float*)d_in[13]);

  // --- 4 stages ---
  float* xa = xyzA; float* xb = xyzB;
  _Float16* feat = featBuf;
  int Nin = 2048, last = 64;
  for (int s = 0; s < 4; s++) {
    int S = Nin / 2, outc = 2 * last;
    int D = last + 3, Dtot = 2 * last + 3, Dpad = ((Dtot + 31) / 32) * 32;
    int sb = 15 + s * 38;
    long Ppre = (long)B * S * K_NEI;

    fps_kernel<<<B, 256, 0, stream>>>(xa, fidx, Nin, S);
    gather_centers<<<B * S, 64, 0, stream>>>(xa, fidx, feat, xb, cfeat, Nin, S, last);
    knn_kernel<<<dim3(S / 128, B), 128, 0, stream>>>(xa, xb, nidx, Nin, S);

    (void)hipMemsetAsync(bsum, 0, 32 * sizeof(float), stream);
    grp_sumsq<<<B * S, 128, 0, stream>>>(feat, xa, nidx, bsum, Nin, S, last, last, D);
    grp_std<<<1, 32, 0, stream>>>(bsum, stdinv, (float)((long)S * K_NEI * D - 1));
    (void)hipMemsetAsync(actG, 0, (size_t)Ppre * Dpad * 2, stream);
    grp_emit<<<B * S, 128, 0, stream>>>(feat, xa, nidx, cfeat,
        (const float*)d_in[sb + 0], (const float*)d_in[sb + 1], stdinv, actG,
        Nin, S, last, last, D, Dpad);

    convbn(Wt[s], actG, actP, nullptr, 1, Ppre, Dpad, outc, outc,
           (const float*)d_in[sb + 34 + 3], (const float*)d_in[sb + 34 + 2]);
    for (int blk = 0; blk < 2; blk++) {
      int i1 = sb + 18 + blk * 8, i2 = i1 + 4;
      convbn(Wpre[s][blk * 2 + 0], actP, actT, nullptr, 1, Ppre, outc, outc, outc,
             (const float*)d_in[i1 + 3], (const float*)d_in[i1 + 2]);
      convbn(Wpre[s][blk * 2 + 1], actT, actP, actP, 1, Ppre, outc, outc, outc,
             (const float*)d_in[i2 + 3], (const float*)d_in[i2 + 2]);
    }
    {
      long tot = (long)B * S * (outc >> 3);
      pool_rows<<<(unsigned)((tot + 255) / 256), 256, 0, stream>>>(
          actP, featBuf, (long)B * S, K_NEI, outc);
    }
    feat = featBuf;
    long Ppos = (long)B * S;
    for (int blk = 0; blk < 2; blk++) {
      int i1 = sb + 2 + blk * 8, i2 = i1 + 4;
      convbn(Wpos[s][blk * 2 + 0], feat, actT, nullptr, 1, Ppos, outc, outc, outc,
             (const float*)d_in[i1 + 3], (const float*)d_in[i1 + 2]);
      convbn(Wpos[s][blk * 2 + 1], actT, feat, feat, 1, Ppos, outc, outc, outc,
             (const float*)d_in[i2 + 3], (const float*)d_in[i2 + 2]);
    }
    float* t = xa; xa = xb; xb = t;
    Nin = S; last = outc;
  }

  // --- classifier head ---
  {
    long tot = (long)B * (1024 >> 3);
    pool_rows<<<(unsigned)((tot + 255) / 256), 256, 0, stream>>>(
        feat, hbuf, (long)B, 128, 1024);
  }
  convbn(Wl1, hbuf, actT, nullptr, 1, (long)B, 1024, 512, 512,
         (const float*)d_in[4], (const float*)d_in[3]);
  convbn(Wl2, actT, actP, nullptr, 1, (long)B, 512, 256, 256,
         (const float*)d_in[8], (const float*)d_in[7]);
  {
    dim3 gg(1, 1);
    wmma_gemm<<<gg, 256, 0, stream>>>(Wl3, actP, Ybuf, (long)B, 256, 64);
    final_bias<<<(32 * 40 + 255) / 256, 256, 0, stream>>>(
        Ybuf, (const float*)d_in[10], (float*)d_out, 64);
  }
}
